// GATEConvGraphGymLayer_84576495992851
// MI455X (gfx1250) — compile-verified, matrix operand order checked by reading, not run on hardware
//
#include <hip/hip_runtime.h>
#include <hip/hip_bf16.h>

// ---------------------------------------------------------------------------
// Types for CDNA5 WMMA (wave32)
// ---------------------------------------------------------------------------
typedef _Float16 v16h __attribute__((ext_vector_type(16)));
typedef _Float16 v8h  __attribute__((ext_vector_type(8)));
typedef float    v8f  __attribute__((ext_vector_type(8)));

#define DIN  128   // input feature dim == H*C
#define HC   128   // H*C
#define HH   4     // heads
#define CC   32    // channels per head
#define DE   32    // edge feature dim

// ---------------------------------------------------------------------------
// f16 16x32 A/B fragment loader per CDNA5 ISA layout:
//   lanes 0-15 : row/col = lane,     halfs[0..7]=K(kb..kb+7),  halfs[8..15]=K(kb+16..kb+23)
//   lanes 16-31: row/col = lane-16,  halfs[0..7]=K(kb+8..+15), halfs[8..15]=K(kb+24..+31)
// base must be row-major with given stride (A: M-major, B: N-major == W^T)
// ---------------------------------------------------------------------------
__device__ inline v16h load_frag16(const _Float16* __restrict__ base, int row,
                                   int stride, int kb, int lane) {
  int koff = (lane & 16) ? 8 : 0;
  const _Float16* p = base + (size_t)row * stride + kb + koff;
  v8h c0 = *(const v8h*)(p);
  v8h c1 = *(const v8h*)(p + 16);
  return __builtin_shufflevector(c0, c1, 0,1,2,3,4,5,6,7,8,9,10,11,12,13,14,15);
}

// ---------------------------------------------------------------------------
// K2: xl = x @ W   (M x 128) * (128 x 128), f16 in, f32 + f16 out
// 8 waves per block; wave w owns M-tile (blockIdx*8+w), all 8 N-tiles.
// ---------------------------------------------------------------------------
__global__ void gemm_xw(const _Float16* __restrict__ Ah,   // [M,128] row-major f16
                        const _Float16* __restrict__ Bt,   // [128,128] = W^T (N-major)
                        float* __restrict__ C32,           // [M,128] f32
                        _Float16* __restrict__ Ch,         // [M,128] f16
                        int M) {
  int lane = threadIdx.x & 31;
  int wave = threadIdx.x >> 5;
  int mt = blockIdx.x * 8 + wave;
  if (mt * 16 >= M) return;

  int row = mt * 16 + (lane & 15);
  bool rowok = row < M;
  int rsafe = rowok ? row : (M - 1);

  v8f acc[8] = {};
  for (int kb = 0; kb < DIN; kb += 32) {
    v16h a = {};
    if (rowok) a = load_frag16(Ah, rsafe, DIN, kb, lane);
#pragma unroll
    for (int t = 0; t < 8; t++) {
      int col = t * 16 + (lane & 15);
      v16h b = load_frag16(Bt, col, DIN, kb, lane);
      acc[t] = __builtin_amdgcn_wmma_f32_16x16x32_f16(
          false, a, false, b, (short)0, acc[t], false, false);
    }
  }
  // D layout: lanes 0-15: VGPR j -> M=j ; lanes 16-31: VGPR j -> M=8+j ; N=lane&15
  int r0 = mt * 16 + ((lane & 16) ? 8 : 0);
  int cb = lane & 15;
#pragma unroll
  for (int t = 0; t < 8; t++) {
#pragma unroll
    for (int j = 0; j < 8; j++) {
      int r = r0 + j;
      if (r < M) {
        float v = acc[t][j];
        C32[(size_t)r * HC + t * 16 + cb] = v;
        Ch[(size_t)r * HC + t * 16 + cb] = (_Float16)v;
      }
    }
  }
}

// ---------------------------------------------------------------------------
// K11: final = (out+bias)_f16 @ proj_W  + proj_b   (M x 128) * (128 x 32)
// ---------------------------------------------------------------------------
__global__ void gemm_proj(const _Float16* __restrict__ Ah,  // [M,128] f16
                          const _Float16* __restrict__ Bt,  // [32,128] = proj_W^T
                          const float* __restrict__ pb,     // [32]
                          float* __restrict__ out,          // [M,32] f32
                          int M) {
  int lane = threadIdx.x & 31;
  int wave = threadIdx.x >> 5;
  int mt = blockIdx.x * 8 + wave;
  if (mt * 16 >= M) return;

  int row = mt * 16 + (lane & 15);
  bool rowok = row < M;
  int rsafe = rowok ? row : (M - 1);

  v8f acc[2] = {};
  for (int kb = 0; kb < DIN; kb += 32) {
    v16h a = {};
    if (rowok) a = load_frag16(Ah, rsafe, DIN, kb, lane);
#pragma unroll
    for (int t = 0; t < 2; t++) {
      int col = t * 16 + (lane & 15);
      v16h b = load_frag16(Bt, col, DIN, kb, lane);
      acc[t] = __builtin_amdgcn_wmma_f32_16x16x32_f16(
          false, a, false, b, (short)0, acc[t], false, false);
    }
  }
  int r0 = mt * 16 + ((lane & 16) ? 8 : 0);
  int cb = lane & 15;
#pragma unroll
  for (int t = 0; t < 2; t++) {
#pragma unroll
    for (int j = 0; j < 8; j++) {
      int r = r0 + j;
      if (r < M) out[(size_t)r * CC + t * 16 + cb] = acc[t][j] + pb[t * 16 + cb];
    }
  }
}

// ---------------------------------------------------------------------------
// K0: fold lin_edge_W with att_edge -> vedge[32,4]; transpose W and proj_W to f16
// ---------------------------------------------------------------------------
__global__ void k_prep(const float* __restrict__ W, const float* __restrict__ linW,
                       const float* __restrict__ att_edge,
                       const float* __restrict__ projW,
                       _Float16* __restrict__ Wt, _Float16* __restrict__ Pt,
                       float* __restrict__ vedge) {
  int tid = threadIdx.x;
  if (tid < DE * HH) {
    int d = tid >> 2, h = tid & 3;
    float s = 0.f;
    for (int c = 0; c < CC; c++) s += linW[d * HC + h * CC + c] * att_edge[h * CC + c];
    vedge[d * HH + h] = s;
  }
  for (int i = tid; i < HC * DIN; i += blockDim.x)           // Wt[n*128+k] = W[k*128+n]
    Wt[i] = (_Float16)W[(i & 127) * HC + (i >> 7)];
  for (int i = tid; i < CC * DIN; i += blockDim.x)           // Pt[n*128+k] = projW[k*32+n]
    Pt[i] = (_Float16)projW[(i & 127) * CC + (i >> 7)];
}

__global__ void k_cvt_f16(const float* __restrict__ in, _Float16* __restrict__ out,
                          size_t n) {
  size_t i = (size_t)blockIdx.x * blockDim.x + threadIdx.x;
  if (i < n) out[i] = (_Float16)in[i];
}

__global__ void k_zero(float* __restrict__ p, size_t n) {
  size_t i = (size_t)blockIdx.x * blockDim.x + threadIdx.x;
  if (i < n) p[i] = 0.f;
}

// ---------------------------------------------------------------------------
// K5: a_src/a_dst per (node, head)
// ---------------------------------------------------------------------------
__global__ void k_attn_node(const float* __restrict__ xl, const float* __restrict__ att_src,
                            const float* __restrict__ att_dst,
                            float* __restrict__ asrc, float* __restrict__ adst, int N) {
  int idx = blockIdx.x * blockDim.x + threadIdx.x;
  if (idx >= N * HH) return;
  int n = idx >> 2, h = idx & 3;
  const float* row = xl + (size_t)n * HC + h * CC;
  float ss = 0.f, sd = 0.f;
  for (int c = 0; c < CC; c++) {
    float v = row[c];
    ss += v * att_src[h * CC + c];
    sd += v * att_dst[h * CC + c];
  }
  asrc[idx] = ss;
  adst[idx] = sd;
}

// ---------------------------------------------------------------------------
// K6: degree + edge_attr segment sum over dst (one thread per (edge, channel))
// ---------------------------------------------------------------------------
__global__ void k_deg_easum(const int* __restrict__ dst, const float* __restrict__ ea,
                            float* __restrict__ deg, float* __restrict__ easum,
                            long long E) {
  long long gid = (long long)blockIdx.x * blockDim.x + threadIdx.x;
  if (gid >= E * DE) return;
  long long e = gid >> 5;
  int c = (int)(gid & 31);
  int d = dst[e];
  atomicAdd(&easum[(size_t)d * DE + c], ea[gid]);
  if (c == 0) atomicAdd(&deg[d], 1.0f);
}

__global__ void k_eamean(float* __restrict__ easum, const float* __restrict__ deg,
                         int N) {
  int i = blockIdx.x * blockDim.x + threadIdx.x;
  if (i < N * DE) easum[i] /= fmaxf(deg[i >> 5], 1.0f);
}

// ordered-uint encoding for float atomic max (handles negatives)
__device__ inline unsigned fkey(float x) {
  unsigned u = __float_as_uint(x);
  return (u & 0x80000000u) ? ~u : (u | 0x80000000u);
}
__device__ inline float fdec(unsigned key) {
  unsigned u = (key & 0x80000000u) ? (key & 0x7FFFFFFFu) : ~key;
  return __uint_as_float(u);
}

// ---------------------------------------------------------------------------
// K8: per-edge attention logits + leaky_relu + segment max (incl. self loops)
// ---------------------------------------------------------------------------
__global__ void k_alpha(const int* __restrict__ src, const int* __restrict__ dst,
                        const float* __restrict__ ea, const float* __restrict__ eamean,
                        const float* __restrict__ vedge,
                        const float* __restrict__ asrc, const float* __restrict__ adst,
                        float* __restrict__ alpha, unsigned* __restrict__ amax,
                        long long E, int N) {
  long long idx = (long long)blockIdx.x * blockDim.x + threadIdx.x;
  if (idx >= E + N) return;
  int s, d;
  const float* row;
  if (idx < E) { s = src[idx]; d = dst[idx]; row = ea + (size_t)idx * DE; }
  else         { s = d = (int)(idx - E);     row = eamean + (size_t)s * DE; }
  float ae0 = 0.f, ae1 = 0.f, ae2 = 0.f, ae3 = 0.f;
  for (int k = 0; k < DE; k++) {
    float v = row[k];
    ae0 += v * vedge[k * 4 + 0];
    ae1 += v * vedge[k * 4 + 1];
    ae2 += v * vedge[k * 4 + 2];
    ae3 += v * vedge[k * 4 + 3];
  }
  float aeh[4] = {ae0, ae1, ae2, ae3};
#pragma unroll
  for (int h = 0; h < 4; h++) {
    float a = asrc[(size_t)s * HH + h] + adst[(size_t)d * HH + h] + aeh[h];
    a = a > 0.f ? a : 0.2f * a;                 // leaky_relu, slope 0.2
    alpha[(size_t)idx * HH + h] = a;
    atomicMax(&amax[(size_t)d * HH + h], fkey(a));
  }
}

// ---------------------------------------------------------------------------
// K10: alpha = exp(alpha - amax[dst]); denom += alpha
// ---------------------------------------------------------------------------
__global__ void k_exp(const int* __restrict__ dst, float* __restrict__ alpha,
                      const unsigned* __restrict__ amax, float* __restrict__ denom,
                      long long E, int N) {
  long long idx = (long long)blockIdx.x * blockDim.x + threadIdx.x;
  if (idx >= E + N) return;
  int d = (idx < E) ? dst[idx] : (int)(idx - E);
#pragma unroll
  for (int h = 0; h < 4; h++) {
    float m = fdec(amax[(size_t)d * HH + h]);
    float a = __expf(alpha[(size_t)idx * HH + h] - m);
    alpha[(size_t)idx * HH + h] = a;
    atomicAdd(&denom[(size_t)d * HH + h], a);
  }
}

// ---------------------------------------------------------------------------
// K11: out[dst] += (alpha/denom) * xl[src]  — 32 lanes per edge (lane = channel)
// ---------------------------------------------------------------------------
__global__ void k_agg(const int* __restrict__ src, const int* __restrict__ dst,
                      const float* __restrict__ alpha, const float* __restrict__ denom,
                      const _Float16* __restrict__ xlh, float* __restrict__ out,
                      long long E, int N) {
  long long gid = (long long)blockIdx.x * blockDim.x + threadIdx.x;
  if (gid >= (E + N) * 32) return;
  long long e = gid >> 5;
  int c = (int)(gid & 31);
  int s, d;
  if (e < E) { s = src[e]; d = dst[e]; }
  else       { s = d = (int)(e - E); }
  const _Float16* srow = xlh + (size_t)s * HC;
  __builtin_prefetch(srow, 0, 1);   // global_prefetch_b8
#pragma unroll
  for (int h = 0; h < 4; h++) {
    float w = alpha[(size_t)e * HH + h] / (denom[(size_t)d * HH + h] + 1e-16f);
    float v = (float)srow[h * CC + c];
    atomicAdd(&out[(size_t)d * HC + h * CC + c], w * v);
  }
}

__global__ void k_bias_cvt(const float* __restrict__ acc, const float* __restrict__ bias,
                           _Float16* __restrict__ outh, size_t n) {
  size_t i = (size_t)blockIdx.x * blockDim.x + threadIdx.x;
  if (i < n) outh[i] = (_Float16)(acc[i] + bias[i & (HC - 1)]);
}

// ---------------------------------------------------------------------------
static inline unsigned cdiv(long long a, long long b) { return (unsigned)((a + b - 1) / b); }

extern "C" void kernel_launch(void* const* d_in, const int* in_sizes, int n_in,
                              void* d_out, int out_size, void* d_ws, size_t ws_size,
                              hipStream_t stream) {
  const float* x        = (const float*)d_in[0];
  const int*   ei       = (const int*)d_in[1];
  const float* edge_attr= (const float*)d_in[2];
  const float* W        = (const float*)d_in[3];
  const float* att_src  = (const float*)d_in[4];
  const float* att_dst  = (const float*)d_in[5];
  const float* linW     = (const float*)d_in[6];
  const float* att_edge = (const float*)d_in[7];
  const float* bias     = (const float*)d_in[8];
  const float* projW    = (const float*)d_in[9];
  const float* projb    = (const float*)d_in[10];
  float* outp = (float*)d_out;

  const int N = in_sizes[0] / DIN;
  const long long E = in_sizes[1] / 2;
  const int* src = ei;
  const int* dst = ei + E;

  // ---- workspace layout (256B aligned slots) ----
  size_t off = 0;
  auto take = [&](size_t bytes) -> char* {
    char* p = (char*)d_ws + off;
    off += (bytes + 255) & ~(size_t)255;
    return p;
  };
  _Float16* xh    = (_Float16*)take((size_t)N * DIN * 2);   // x f16; reused for (out+bias) f16
  _Float16* Wt    = (_Float16*)take((size_t)HC * DIN * 2);
  _Float16* Pt    = (_Float16*)take((size_t)CC * DIN * 2);
  float*    vedge = (float*)take(DE * HH * 4);
  float*    xl32  = (float*)take((size_t)N * HC * 4);       // xl f32; reused as out accumulator
  _Float16* xlh   = (_Float16*)take((size_t)N * HC * 2);
  float*    asrc  = (float*)take((size_t)N * HH * 4);
  float*    adst  = (float*)take((size_t)N * HH * 4);
  float*    deg   = (float*)take((size_t)N * 4);
  float*    easum = (float*)take((size_t)N * DE * 4);       // becomes ea_mean in place
  float*    alpha = (float*)take((size_t)(E + N) * HH * 4);
  unsigned* amax  = (unsigned*)take((size_t)N * HH * 4);
  float*    denom = (float*)take((size_t)N * HH * 4);
  (void)ws_size;

  const int B = 256;
  const long long Etot = E + N;
  const unsigned mblk = cdiv((N + 15) / 16, 8);             // GEMM blocks (8 waves/blk)

  // 0) fold weights + transposes
  k_prep<<<1, B, 0, stream>>>(W, linW, att_edge, projW, Wt, Pt, vedge);
  // 1) x -> f16
  k_cvt_f16<<<cdiv((long long)N * DIN, B), B, 0, stream>>>(x, xh, (size_t)N * DIN);
  // 2) zero accumulators
  k_zero<<<cdiv(N, B), B, 0, stream>>>(deg, (size_t)N);
  k_zero<<<cdiv((long long)N * DE, B), B, 0, stream>>>(easum, (size_t)N * DE);
  k_zero<<<cdiv((long long)N * HH, B), B, 0, stream>>>((float*)amax, (size_t)N * HH);
  k_zero<<<cdiv((long long)N * HH, B), B, 0, stream>>>(denom, (size_t)N * HH);
  // 3) xl = x @ W   (WMMA)
  gemm_xw<<<mblk, B, 0, stream>>>(xh, Wt, xl32, xlh, N);
  // 4) a_src / a_dst
  k_attn_node<<<cdiv((long long)N * HH, B), B, 0, stream>>>(xl32, att_src, att_dst,
                                                            asrc, adst, N);
  // 5) degree + edge_attr mean over dst
  k_deg_easum<<<cdiv(E * DE, B), B, 0, stream>>>(dst, edge_attr, deg, easum, E);
  k_eamean<<<cdiv((long long)N * DE, B), B, 0, stream>>>(easum, deg, N);
  // 6) attention logits + segment max
  k_alpha<<<cdiv(Etot, B), B, 0, stream>>>(src, dst, edge_attr, easum, vedge,
                                           asrc, adst, alpha, amax, E, N);
  // 7) reuse xl32 as output accumulator (xl f16 copy used for gathers)
  k_zero<<<cdiv((long long)N * HC, B), B, 0, stream>>>(xl32, (size_t)N * HC);
  // 8) softmax numerator + denominator
  k_exp<<<cdiv(Etot, B), B, 0, stream>>>(dst, alpha, amax, denom, E, N);
  // 9) weighted scatter-add aggregation
  k_agg<<<cdiv(Etot * 32, B), B, 0, stream>>>(src, dst, alpha, denom, xlh, xl32, E, N);
  // 10) + bias, convert to f16 (reuse xh)
  k_bias_cvt<<<cdiv((long long)N * HC, B), B, 0, stream>>>(xl32, bias, xh, (size_t)N * HC);
  // 11) final projection (WMMA) + proj_b -> d_out
  gemm_proj<<<mblk, B, 0, stream>>>(xh, Pt, projb, outp, N);
}
